// Merger_49289044689057
// MI455X (gfx1250) — matrix operand order repaired and checked by reading, not compiled, not run on hardware
//
#include <hip/hip_runtime.h>
#include <hip/hip_bf16.h>

typedef __attribute__((ext_vector_type(16))) _Float16 v16h;
typedef __attribute__((ext_vector_type(8)))  float    v8f;

#define BATCH 2
#define CAM   4
#define CCH   14                  // feature channels
#define XCH   ((CCH + 1) * CAM)   // 60 stacked channels
#define Hh    480
#define Ww    640
#define HW    (Hh * Ww)
#define SCALE 10.0f

// Workspace layout (int32 units): [0,192) float M matrices (t,s) 12 floats each,
// maps start at int offset MAP_OFF (1 KB aligned region).
#define MAP_OFF 256

// ---------------------------------------------------------------------------
// M(t,s) = inv(T_t) @ T_s for all (t,s). T are rigid [R|t]: inv = [R^T|-R^T t].
// ---------------------------------------------------------------------------
__global__ void setup_M(const float* __restrict__ T, float* __restrict__ wsM) {
    int id = threadIdx.x;
    if (id >= CAM * CAM) return;
    int t = id / CAM, s = id % CAM;
    const float* Tt = T + t * 16;
    const float* Ts = T + s * 16;
    float A[3][4];
    for (int i = 0; i < 3; ++i)
        for (int k = 0; k < 3; ++k)
            A[i][k] = Tt[k * 4 + i];               // R^T
    for (int i = 0; i < 3; ++i)
        A[i][3] = -(A[i][0] * Tt[0 * 4 + 3] + A[i][1] * Tt[1 * 4 + 3] +
                    A[i][2] * Tt[2 * 4 + 3]);       // -R^T t
    float* M = wsM + id * 12;
    for (int i = 0; i < 3; ++i)
        for (int j = 0; j < 4; ++j) {
            float v = A[i][3] * Ts[3 * 4 + j];
            for (int k = 0; k < 3; ++k) v += A[i][k] * Ts[k * 4 + j];
            M[i * 4 + j] = v;
        }
}

__global__ void init_map(int* __restrict__ maps, int count) {
    int i = blockIdx.x * blockDim.x + threadIdx.x;
    if (i < count) maps[i] = -1;
}

// ---------------------------------------------------------------------------
// For each (b, t, s!=t, source pixel n): transform xyz into frame t, project
// with K_t, round-to-nearest-even, scatter the SOURCE PIXEL INDEX. The
// reference's row flip (dst[:, :, ::-1, :]) is folded in via row (H-1-y).
// ---------------------------------------------------------------------------
__global__ void scatter_map(const float* __restrict__ xyzi,
                            const float* __restrict__ K,
                            const float* __restrict__ wsM,
                            int* __restrict__ maps) {
    long long gid = (long long)blockIdx.x * blockDim.x + threadIdx.x;
    const long long total = (long long)BATCH * CAM * CAM * HW;
    if (gid >= total) return;
    int n    = (int)(gid % HW);
    int rest = (int)(gid / HW);
    int s = rest % CAM; rest /= CAM;
    int t = rest % CAM;
    int b = rest / CAM;
    if (s == t) return;

    const float* M = wsM + (t * CAM + s) * 12;
    int base = (b * CAM + s) * 4 * HW;
    float X = xyzi[base + 0 * HW + n];
    float Y = xyzi[base + 1 * HW + n];
    float Z = xyzi[base + 2 * HW + n];
    float Xp = M[0] * X + M[1] * Y + M[2]  * Z + M[3];
    float Yp = M[4] * X + M[5] * Y + M[6]  * Z + M[7];
    float Zp = M[8] * X + M[9] * Y + M[10] * Z + M[11];

    const float* Kt = K + t * 9;
    float invz = 1.0f / Zp;
    float px = rintf(Kt[0] * Xp * invz + Kt[1] * Yp * invz + Kt[2]);
    float py = rintf(Kt[3] * Xp * invz + Kt[4] * Yp * invz + Kt[5]);
    if (px > 0.0f && px < (float)Ww && py > 0.0f && py < (float)Hh) {
        int ix = (int)px, iy = (int)py;
        maps[((b * CAM + t) * CAM + s) * HW + (Hh - 1 - iy) * Ww + ix] = n;
    }
}

// ---------------------------------------------------------------------------
// Gather one cam-group's 15 channels (14 feature + z) for target pixel n into
// one 16-element f16 B-chunk (slot 15 = K padding). One map lookup per group;
// channel offsets are compile-time constants.
// ---------------------------------------------------------------------------
__device__ __forceinline__ v16h gather_group(int g, int b, int t, int n,
                                             const float* __restrict__ feature,
                                             const float* __restrict__ xyzi,
                                             const int* __restrict__ maps) {
    // order = [t] + [s for s in range(CAM) if s != t]
    int s = (g == 0) ? t : ((g - 1 < t) ? (g - 1) : g);
    int src;
    bool valid;
    if (g == 0) {                       // own camera: identity mapping
        src = n;
        valid = true;
    } else {                            // projected camera: one map lookup
        src = maps[((b * CAM + t) * CAM + s) * HW + n];
        valid = (src >= 0);
        src = valid ? src : 0;          // safe address for masked-off loads
    }
    const float* fb = feature + (b * CAM + s) * CCH * HW + src;
    float zv = xyzi[((b * CAM + s) * 4 + 2) * HW + src];
    v16h r;
    for (int cc = 0; cc < CCH; ++cc) {
        float v = fb[cc * HW];          // constant channel offset
        r[cc] = (_Float16)(valid ? v : 0.0f);
    }
    r[14] = (_Float16)(valid ? zv : 0.0f);
    r[15] = (_Float16)0.0f;
    return r;
}

// ---------------------------------------------------------------------------
// Per-pixel 14x60 linear layer via v_wmma_f32_16x16x32_f16, K permuted so
// slot 16g+cc holds channel g*15+cc (slot 16g+15 = zero pad). With this code's contiguous
// 16-bit B layout (lane half covers K = 16*half + e), each lane's chunk0 is
// exactly cam-group `half` and chunk1 is group `2+half` -> no /15, <=2 map
// loads per lane per tile. A applies the same permutation, built once per
// wave; wave-uniform grid-stride loop keeps EXEC all-ones at the WMMAs.
// ---------------------------------------------------------------------------
__global__ void __launch_bounds__(256)
gemm_wmma(const float* __restrict__ feature, const float* __restrict__ xyzi,
          const float* __restrict__ unet_w, const float* __restrict__ unet_b,
          const int* __restrict__ maps, float* __restrict__ out) {
    const int lane = threadIdx.x & 31;
    const int half = lane >> 4;
    const int col  = lane & 15;
    const int wave   = blockIdx.x * (blockDim.x >> 5) + (threadIdx.x >> 5);
    const int nwaves = gridDim.x * (blockDim.x >> 5);

    // A operand (16x32 f16 per chunk), contiguous K order (matches B): lane (half, m=lane&15)
    // holds K-slot 16*half + e (+32 for chunk1). Map slot ->
    // permuted channel g*15+cc and fold in 1/SCALE.
    const int m = lane & 15;
    v16h a0, a1;
    for (int e = 0; e < 16; ++e) {
        int slot0 = 16 * half + e;
        int slot1 = slot0 + 32;
        int g0 = slot0 >> 4, c0 = slot0 & 15;
        int g1 = slot1 >> 4, c1 = slot1 & 15;
        float w0 = (m < CCH && c0 < 15) ? unet_w[m * XCH + g0 * 15 + c0] * (1.0f / SCALE) : 0.0f;
        float w1 = (m < CCH && c1 < 15) ? unet_w[m * XCH + g1 * 15 + c1] * (1.0f / SCALE) : 0.0f;
        a0[e] = (_Float16)w0;
        a1[e] = (_Float16)w1;
    }
    float bias[8];
    for (int r = 0; r < 8; ++r) {
        int o = r + 8 * half;
        bias[r] = (o < CCH) ? unet_b[o] : 0.0f;
    }

    const int tiles_per_img = HW / 16;
    const int ntiles = BATCH * CAM * tiles_per_img;
    for (int tile = wave; tile < ntiles; tile += nwaves) {
        int ptile = tile % tiles_per_img;
        int bt    = tile / tiles_per_img;
        int t = bt % CAM, b = bt / CAM;
        int n = ptile * 16 + col;

        // B operand: lane (half, col) chunk0 slots = 16*half+e -> group half;
        // chunk1 slots = 32+16*half+e -> group 2+half.
        v16h b0 = gather_group(half,     b, t, n, feature, xyzi, maps);
        v16h b1 = gather_group(2 + half, b, t, n, feature, xyzi, maps);

        v8f acc = {};
        acc = __builtin_amdgcn_wmma_f32_16x16x32_f16(false, a0, false, b0,
                                                     (short)0, acc, false, false);
        acc = __builtin_amdgcn_wmma_f32_16x16x32_f16(false, a1, false, b1,
                                                     (short)0, acc, false, false);

        // D layout: lane (half, col), VGPR r -> (o = r + 8*half, pixel n).
        int obase = (b * CAM + t) * CCH * HW + n;
        for (int r = 0; r < 8; ++r) {
            int o = r + 8 * half;
            if (o < CCH)
                out[obase + o * HW] = acc[r] + bias[r];
        }
    }
}

__global__ void copy_label(const int* __restrict__ label, int* __restrict__ dst,
                           int count) {
    int i = blockIdx.x * blockDim.x + threadIdx.x;
    if (i < count) dst[i] = label[i];
}

extern "C" void kernel_launch(void* const* d_in, const int* in_sizes, int n_in,
                              void* d_out, int out_size, void* d_ws, size_t ws_size,
                              hipStream_t stream) {
    const float* feature = (const float*)d_in[0];
    const float* xyzi    = (const float*)d_in[1];
    const float* T       = (const float*)d_in[2];
    const float* K       = (const float*)d_in[3];
    const float* unet_w  = (const float*)d_in[4];
    const float* unet_b  = (const float*)d_in[5];
    const int*   label   = (const int*)d_in[6];
    float* out = (float*)d_out;

    float* wsM  = (float*)d_ws;
    int*   maps = (int*)d_ws + MAP_OFF;

    setup_M<<<1, 32, 0, stream>>>(T, wsM);

    const long long mapN = (long long)BATCH * CAM * CAM * HW;   // 9,830,400
    init_map<<<(int)((mapN + 255) / 256), 256, 0, stream>>>(maps, (int)mapN);
    scatter_map<<<(int)((mapN + 255) / 256), 256, 0, stream>>>(xyzi, K, wsM, maps);

    gemm_wmma<<<2048, 256, 0, stream>>>(feature, xyzi, unet_w, unet_b, maps, out);

    const long long nsoft = (long long)BATCH * CAM * CCH * HW;  // 34,406,400 floats
    int lcount = in_sizes[6];                                    // label elements
    copy_label<<<(lcount + 255) / 256, 256, 0, stream>>>(label, (int*)(out + nsoft),
                                                         lcount);
}
